// Attention_2396591751951
// MI455X (gfx1250) — compile-verified, matrix operand order checked by reading, not run on hardware
//
#include <hip/hip_runtime.h>

// ---------------------------------------------------------------------------
// Talking-heads attention for MI455X (gfx1250), bf16 WMMA pipeline.
//   1) cast f32 -> bf16 (x, Wq, Wkv, Wo)
//   2) GEMM q  = x @ Wq      (bf16 in, f32 acc, bf16 out)
//   3) GEMM kv = x @ Wkv
//   4) fused talking-heads attention (two-pass streaming softmax)
//   5) GEMM out = heads @ Wo + bo   (f32 out)
// GEMM tiles are staged global->LDS with CDNA5 async-to-LDS ops when the
// toolchain exposes them (ASYNCcnt path), else vectorized b128 copies.
// ---------------------------------------------------------------------------

typedef unsigned short bf16_t;
typedef __attribute__((ext_vector_type(16))) __bf16 v16bf;
typedef __attribute__((ext_vector_type(8)))  float  v8f;
typedef __attribute__((ext_vector_type(8)))  unsigned short bfx8;  // 16 bytes
typedef int v4i_vs __attribute__((__vector_size__(4 * sizeof(int))));

#define DIMM      1024
#define NHEADS    16
#define DHEAD     64
#define SEQN      1024
#define NBATCH    4
#define MROWS     4096          // b * n
#define ATT_SCALE 0.125f        // 64^-0.5
#define JB        64            // j-block width in attention

#define AS1 __attribute__((address_space(1)))
#define AS3 __attribute__((address_space(3)))

typedef AS1 v4i_vs* gv4_ptr;   // global 16B-vector pointer (64-bit)
typedef AS3 v4i_vs* lv4_ptr;   // LDS 16B-vector pointer (32-bit)

#if defined(__has_builtin)
#  if __has_builtin(__builtin_amdgcn_global_load_async_to_lds_b128)
#    define USE_ASYNC_LDS 1
#  endif
#  if __has_builtin(__builtin_amdgcn_s_wait_asynccnt)
#    define HAVE_WAIT_ASYNC_BUILTIN 1
#  endif
#endif

__device__ __forceinline__ void wait_asynccnt0() {
#if defined(HAVE_WAIT_ASYNC_BUILTIN)
  __builtin_amdgcn_s_wait_asynccnt(0);
#else
  asm volatile("s_wait_asynccnt 0x0" ::: "memory");
#endif
}

// generic -> AS1 / AS3 pointer bridges (via integers; LDS generic addresses
// carry the LDS byte offset in their low 32 bits per the aperture rules)
__device__ __forceinline__ gv4_ptr to_global_v4(const void* p) {
  return (gv4_ptr)(unsigned long long)p;
}
__device__ __forceinline__ lv4_ptr to_lds_v4(const void* p) {
  return (lv4_ptr)(unsigned)(unsigned long long)p;
}

// ---------------- bf16 helpers (raw-bits storage) ----------------
__device__ __forceinline__ bf16_t f2bf(float f) {
  union { float f; unsigned u; } v; v.f = f;
  unsigned r = v.u + 0x7FFFu + ((v.u >> 16) & 1u);   // round-to-nearest-even
  return (bf16_t)(r >> 16);
}
__device__ __forceinline__ float bf2f(bf16_t b) {
  union { unsigned u; float f; } v; v.u = ((unsigned)b) << 16;
  return v.f;
}
__device__ __forceinline__ __bf16 bfbits(bf16_t b) {
  return __builtin_bit_cast(__bf16, b);
}

// K index base for 16-bit WMMA operand fragments (ISA 7.12.2):
//   VGPR pair p (0..7): K = (p<4?0:16) + (p&3)*2 + (lane<16?0:8)
__device__ __forceinline__ int wmma_kbase(int p, int lane) {
  return ((p & 4) << 2) + ((p & 3) << 1) + ((lane & 16) >> 1);
}

__device__ __forceinline__ v8f wmma_bf16(const v16bf& a, const v16bf& b, const v8f& c) {
  return __builtin_amdgcn_wmma_f32_16x16x32_bf16(false, a, false, b, (short)0, c,
                                                 false, false);
}

// ---------------------------------------------------------------------------
// f32 -> bf16 cast, grid-stride
// ---------------------------------------------------------------------------
__global__ void cast_f32_to_bf16(const float* __restrict__ in,
                                 bf16_t* __restrict__ out, int n) {
  int i = blockIdx.x * blockDim.x + threadIdx.x;
  int stride = gridDim.x * blockDim.x;
  for (; i < n; i += stride) out[i] = f2bf(in[i]);
}

// ---------------------------------------------------------------------------
// Generic bf16 GEMM: C[M,N] = A[M,K] @ B[K,N]  (row-major)
// 256 threads = 8 waves; workgroup tile 128x128; wave tile 64x32.
// OUT_F32 ? (outF = acc + bias) : (outB = bf16(acc))
// ---------------------------------------------------------------------------
template <bool OUT_F32>
__global__ __launch_bounds__(256)
void gemm_bf16(const bf16_t* __restrict__ A, const bf16_t* __restrict__ B,
               bf16_t* __restrict__ outB, float* __restrict__ outF,
               const float* __restrict__ bias, int M, int N, int K) {
  __shared__ bf16_t As[128][32];   // 8 KB
  __shared__ bf16_t Bs[32][128];   // 8 KB

  const int tid  = threadIdx.x;
  const int lane = tid & 31;
  const int w    = tid >> 5;
  const int wr   = w >> 2;         // 0..1 -> M offset 0/64
  const int wc   = w & 3;          // 0..3 -> N offset 0/32/64/96
  const int m0   = blockIdx.y * 128;
  const int n0   = blockIdx.x * 128;

  const v8f vzero = {0.f, 0.f, 0.f, 0.f, 0.f, 0.f, 0.f, 0.f};
  v8f acc[4][2];
#pragma unroll
  for (int i = 0; i < 4; ++i)
#pragma unroll
    for (int j = 0; j < 2; ++j) acc[i][j] = vzero;

  for (int k0 = 0; k0 < K; k0 += 32) {
    // ---- stage 128x32 A-tile and 32x128 B-tile, 16B per slot ----
#pragma unroll
    for (int it = 0; it < 2; ++it) {
      int slot = tid + it * 256;                 // 0..511
      int ra = slot >> 2, ca = (slot & 3) * 8;   // A: 4 slots per row
      int rb = slot >> 4, cb = (slot & 15) * 8;  // B: 16 slots per row
      const bf16_t* gA = &A[(size_t)(m0 + ra) * K + (k0 + ca)];
      const bf16_t* gB = &B[(size_t)(k0 + rb) * N + (n0 + cb)];
#if defined(USE_ASYNC_LDS)
      __builtin_amdgcn_global_load_async_to_lds_b128(
          to_global_v4(gA), to_lds_v4(&As[ra][ca]), 0, 0);
      __builtin_amdgcn_global_load_async_to_lds_b128(
          to_global_v4(gB), to_lds_v4(&Bs[rb][cb]), 0, 0);
#else
      *(bfx8*)&As[ra][ca] = *(const bfx8*)gA;
      *(bfx8*)&Bs[rb][cb] = *(const bfx8*)gB;
#endif
      // prefetch next K-step's tiles toward L2
      if (k0 + 32 < K) {
        __builtin_prefetch(gA + 32, 0, 1);
        __builtin_prefetch(gB + (size_t)32 * N, 0, 1);
      }
    }
#if defined(USE_ASYNC_LDS)
    wait_asynccnt0();
#endif
    __syncthreads();

    // ---- fragments + 8 WMMAs ----
    v16bf afr[4], bfr[2];
#pragma unroll
    for (int ms = 0; ms < 4; ++ms) {
      int row = wr * 64 + ms * 16 + (lane & 15);
#pragma unroll
      for (int p = 0; p < 8; ++p) {
        int kb = wmma_kbase(p, lane);
        afr[ms][2 * p]     = bfbits(As[row][kb]);
        afr[ms][2 * p + 1] = bfbits(As[row][kb + 1]);
      }
    }
#pragma unroll
    for (int ns = 0; ns < 2; ++ns) {
      int col = wc * 32 + ns * 16 + (lane & 15);
#pragma unroll
      for (int p = 0; p < 8; ++p) {
        int kb = wmma_kbase(p, lane);
        bfr[ns][2 * p]     = bfbits(Bs[kb][col]);
        bfr[ns][2 * p + 1] = bfbits(Bs[kb + 1][col]);
      }
    }
#pragma unroll
    for (int ms = 0; ms < 4; ++ms)
#pragma unroll
      for (int ns = 0; ns < 2; ++ns)
        acc[ms][ns] = wmma_bf16(afr[ms], bfr[ns], acc[ms][ns]);
    __syncthreads();
  }

  // write out per C-layout: VGPR r -> M = r (+8 for upper half-wave); N = lane&15
#pragma unroll
  for (int ms = 0; ms < 4; ++ms)
#pragma unroll
    for (int ns = 0; ns < 2; ++ns)
#pragma unroll
      for (int r = 0; r < 8; ++r) {
        int grow = m0 + wr * 64 + ms * 16 + r + ((lane & 16) >> 1);
        int gcol = n0 + wc * 32 + ns * 16 + (lane & 15);
        float val = acc[ms][ns][r];
        if constexpr (OUT_F32) {
          outF[(size_t)grow * N + gcol] = val + bias[gcol];
        } else {
          outB[(size_t)grow * N + gcol] = f2bf(val);
        }
      }
}

// ---------------------------------------------------------------------------
// Fused talking-heads attention.
// Grid: (n/16, b). Block: 512 threads = 16 waves; wave w handles head w.
// Two passes over j in 64-wide blocks:
//   pass1: per-head dots -> LDS -> mix_pre -> streaming row max/sum
//   pass2: exact softmax probs -> LDS -> mix_post built directly in
//          A-fragment layout -> WMMA with v -> out accumulator.
// ---------------------------------------------------------------------------
__device__ __forceinline__ void compute_dots(const bf16_t* __restrict__ kvlin,
                                             const v16bf qf[2], int b, int j0,
                                             int head, int lane, v8f dacc[4]) {
  const v8f vzero = {0.f, 0.f, 0.f, 0.f, 0.f, 0.f, 0.f, 0.f};
#pragma unroll
  for (int t = 0; t < 4; ++t) dacc[t] = vzero;
#pragma unroll
  for (int f = 0; f < 2; ++f) {
    v16bf kf[4];
#pragma unroll
    for (int t = 0; t < 4; ++t) {
#pragma unroll
      for (int p = 0; p < 8; ++p) {
        int kb = f * 32 + wmma_kbase(p, lane);      // d offset
        int j  = j0 + t * 16 + (lane & 15);         // key index
        size_t base = ((size_t)(b * SEQN + j)) * (2 * DIMM) + head * DHEAD + kb;
        kf[t][2 * p]     = bfbits(kvlin[base]);
        kf[t][2 * p + 1] = bfbits(kvlin[base + 1]);
      }
    }
#pragma unroll
    for (int t = 0; t < 4; ++t) dacc[t] = wmma_bf16(qf[f], kf[t], dacc[t]);
  }
}

__global__ __launch_bounds__(512)
void talking_heads_attn(const bf16_t* __restrict__ qlin,
                        const bf16_t* __restrict__ kvlin,
                        const float* __restrict__ mixpre,
                        const float* __restrict__ mixpost,
                        bf16_t* __restrict__ outh) {
  __shared__ float lds_dots[NHEADS][16][JB];  // 64 KB, reused for probs in pass 2

  const int tid  = threadIdx.x;
  const int lane = tid & 31;
  const int w    = tid >> 5;          // head id / mixed-head id for this wave
  const int i0   = blockIdx.x * 16;   // query row block
  const int b    = blockIdx.y;

  const int row_i   = lane & 15;            // query row owned for mixing
  const int halfsel = (lane >> 4) & 1;      // j-half: 0 -> 0..31, 1 -> 32..63
  const int hup     = (lane & 16) >> 1;     // +8 M offset for upper half-wave

  // mixing columns for this wave (uniform per wave -> scalarized loads)
  float mp[NHEADS], mq[NHEADS];
#pragma unroll
  for (int h = 0; h < NHEADS; ++h) {
    mp[h] = mixpre[h * NHEADS + w];
    mq[h] = mixpost[h * NHEADS + w];
  }

  // persistent q A-fragments (16 rows x 64 d) for head w
  v16bf qf[2];
#pragma unroll
  for (int f = 0; f < 2; ++f) {
#pragma unroll
    for (int p = 0; p < 8; ++p) {
      int kb = f * 32 + wmma_kbase(p, lane);
      size_t base =
          ((size_t)(b * SEQN + i0 + (lane & 15))) * DIMM + w * DHEAD + kb;
      qf[f][2 * p]     = bfbits(qlin[base]);
      qf[f][2 * p + 1] = bfbits(qlin[base + 1]);
    }
  }

  float m_run = -1e30f, l_run = 0.f;

  // ============================ PASS 1 ============================
  for (int j0 = 0; j0 < SEQN; j0 += JB) {
    v8f dacc[4];
    compute_dots(kvlin, qf, b, j0, w, lane, dacc);
#pragma unroll
    for (int t = 0; t < 4; ++t)
#pragma unroll
      for (int r = 0; r < 8; ++r)
        lds_dots[w][r + hup][t * 16 + (lane & 15)] = dacc[t][r];
    __syncthreads();

    float sv[32];
    float lmax = -1e30f;
#pragma unroll
    for (int jj = 0; jj < 32; ++jj) {
      int j = halfsel * 32 + jj;
      float s = 0.f;
#pragma unroll
      for (int h = 0; h < NHEADS; ++h) s += lds_dots[h][row_i][j] * mp[h];
      s *= ATT_SCALE;
      sv[jj] = s;
      lmax = fmaxf(lmax, s);
    }
    float bmax = fmaxf(lmax, __shfl_xor(lmax, 16, 32));
    float mnew = fmaxf(m_run, bmax);
    float psum = 0.f;
#pragma unroll
    for (int jj = 0; jj < 32; ++jj) psum += __expf(sv[jj] - mnew);
    psum += __shfl_xor(psum, 16, 32);
    l_run = l_run * __expf(m_run - mnew) + psum;
    m_run = mnew;
    __syncthreads();
  }
  const float inv_l = 1.0f / l_run;

  // ============================ PASS 2 ============================
  const v8f vzero = {0.f, 0.f, 0.f, 0.f, 0.f, 0.f, 0.f, 0.f};
  v8f oacc[4];
#pragma unroll
  for (int t = 0; t < 4; ++t) oacc[t] = vzero;

  for (int j0 = 0; j0 < SEQN; j0 += JB) {
    v8f dacc[4];
    compute_dots(kvlin, qf, b, j0, w, lane, dacc);
#pragma unroll
    for (int t = 0; t < 4; ++t)
#pragma unroll
      for (int r = 0; r < 8; ++r)
        lds_dots[w][r + hup][t * 16 + (lane & 15)] = dacc[t][r];
    __syncthreads();

    // exact normalized probabilities for mixed head w
    float pv[32];
#pragma unroll
    for (int jj = 0; jj < 32; ++jj) {
      int j = halfsel * 32 + jj;
      float s = 0.f;
#pragma unroll
      for (int h = 0; h < NHEADS; ++h) s += lds_dots[h][row_i][j] * mp[h];
      s *= ATT_SCALE;
      pv[jj] = __expf(s - m_run) * inv_l;
    }
    __syncthreads();  // everyone done reading dots
#pragma unroll
    for (int jj = 0; jj < 32; ++jj)
      lds_dots[w][row_i][halfsel * 32 + jj] = pv[jj];
    __syncthreads();  // probs visible to all waves

    // build post-mixed attn A-fragments for head w, directly in WMMA layout
    v16bf af[2];
#pragma unroll
    for (int f = 0; f < 2; ++f) {
#pragma unroll
      for (int p = 0; p < 8; ++p) {
        int kb = f * 32 + wmma_kbase(p, lane);  // j offset within block
        int ir = lane & 15;
        float a0 = 0.f, a1 = 0.f;
#pragma unroll
        for (int h = 0; h < NHEADS; ++h) {
          float m = mq[h];
          a0 += lds_dots[h][ir][kb] * m;
          a1 += lds_dots[h][ir][kb + 1] * m;
        }
        af[f][2 * p]     = bfbits(f2bf(a0));
        af[f][2 * p + 1] = bfbits(f2bf(a1));
      }
    }

    // accumulate attn_post @ v with WMMA
#pragma unroll
    for (int f = 0; f < 2; ++f) {
      v16bf vf[4];
#pragma unroll
      for (int t = 0; t < 4; ++t) {
#pragma unroll
        for (int p = 0; p < 8; ++p) {
          int kb = wmma_kbase(p, lane);
          int j  = j0 + f * 32 + kb;
          int d  = t * 16 + (lane & 15);
          size_t base =
              ((size_t)(b * SEQN + j)) * (2 * DIMM) + DIMM + w * DHEAD + d;
          vf[t][2 * p]     = bfbits(kvlin[base]);
          vf[t][2 * p + 1] = bfbits(kvlin[base + 2 * DIMM]);  // j+1
        }
      }
#pragma unroll
      for (int t = 0; t < 4; ++t) oacc[t] = wmma_bf16(af[f], vf[t], oacc[t]);
    }
    __syncthreads();
  }

  // store merged-head output [b, i, w*64 + d] as bf16
#pragma unroll
  for (int t = 0; t < 4; ++t)
#pragma unroll
    for (int r = 0; r < 8; ++r) {
      int irow = i0 + r + hup;
      int d    = t * 16 + (lane & 15);
      outh[((size_t)(b * SEQN + irow)) * DIMM + w * DHEAD + d] =
          f2bf(oacc[t][r]);
    }
}

// ---------------------------------------------------------------------------
// Host-side launcher
// ---------------------------------------------------------------------------
extern "C" void kernel_launch(void* const* d_in, const int* in_sizes, int n_in,
                              void* d_out, int out_size, void* d_ws,
                              size_t ws_size, hipStream_t stream) {
  (void)in_sizes; (void)n_in; (void)out_size; (void)ws_size;
  const float* x       = (const float*)d_in[0];
  const float* Wq      = (const float*)d_in[1];
  const float* Wkv     = (const float*)d_in[2];
  const float* mixpre  = (const float*)d_in[3];
  const float* mixpost = (const float*)d_in[4];
  const float* Wo      = (const float*)d_in[5];
  const float* bo      = (const float*)d_in[6];
  float* out = (float*)d_out;

  char* ws = (char*)d_ws;
  size_t off = 0;
  auto alloc = [&](size_t bytes) -> void* {
    void* p = ws + off;
    off += (bytes + 255) & ~(size_t)255;
    return p;
  };
  bf16_t* xb    = (bf16_t*)alloc((size_t)MROWS * DIMM * 2);      // 8 MB
  bf16_t* wqb   = (bf16_t*)alloc((size_t)DIMM * DIMM * 2);       // 2 MB
  bf16_t* wkvb  = (bf16_t*)alloc((size_t)DIMM * 2 * DIMM * 2);   // 4 MB
  bf16_t* wob   = (bf16_t*)alloc((size_t)DIMM * DIMM * 2);       // 2 MB
  bf16_t* qlin  = (bf16_t*)alloc((size_t)MROWS * DIMM * 2);      // 8 MB
  bf16_t* kvlin = (bf16_t*)alloc((size_t)MROWS * 2 * DIMM * 2);  // 16 MB
  bf16_t* oh    = (bf16_t*)alloc((size_t)MROWS * DIMM * 2);      // 8 MB

  cast_f32_to_bf16<<<512, 256, 0, stream>>>(x, xb, MROWS * DIMM);
  cast_f32_to_bf16<<<256, 256, 0, stream>>>(Wq, wqb, DIMM * DIMM);
  cast_f32_to_bf16<<<256, 256, 0, stream>>>(Wkv, wkvb, DIMM * 2 * DIMM);
  cast_f32_to_bf16<<<256, 256, 0, stream>>>(Wo, wob, DIMM * DIMM);

  // q = x @ Wq   -> qlin [4096,1024] bf16
  gemm_bf16<false><<<dim3(DIMM / 128, MROWS / 128), 256, 0, stream>>>(
      xb, wqb, qlin, nullptr, nullptr, MROWS, DIMM, DIMM);
  // kv = x @ Wkv -> kvlin [4096,2048] bf16 (k cols 0..1023, v cols 1024..2047)
  gemm_bf16<false><<<dim3(2 * DIMM / 128, MROWS / 128), 256, 0, stream>>>(
      xb, wkvb, kvlin, nullptr, nullptr, MROWS, 2 * DIMM, DIMM);

  // fused talking-heads attention -> oh [4096,1024] bf16 (heads merged)
  talking_heads_attn<<<dim3(SEQN / 16, NBATCH), 512, 0, stream>>>(
      qlin, kvlin, mixpre, mixpost, oh);

  // out = oh @ Wo + bo -> f32
  gemm_bf16<true><<<dim3(DIMM / 128, MROWS / 128), 256, 0, stream>>>(
      oh, wob, nullptr, out, bo, MROWS, DIMM, DIMM);
}
